// CausalSelfAttention_42296837931621
// MI455X (gfx1250) — matrix operand order repaired and checked by reading, not compiled, not run on hardware
//
#include <hip/hip_runtime.h>

typedef unsigned short u16;
typedef unsigned int   u32;
typedef unsigned long long u64;

typedef __attribute__((ext_vector_type(16))) __bf16 v16bf;
typedef __attribute__((ext_vector_type(8)))  __bf16 v8bf;
typedef __attribute__((ext_vector_type(8)))  float  v8f;
typedef __attribute__((ext_vector_type(4)))  u32    v4u;

union V8  { v4u u; v8bf b; u16 s[8]; };
union V16 { v16bf v; v8bf h[2]; };

__device__ __forceinline__ u16 f2bf(float f) {
  union { float f; u32 u; } c; c.f = f;
  u32 u = c.u;
  u32 r = (u + 0x7FFFu + ((u >> 16) & 1u)) >> 16;
  return (u16)r;
}

__device__ __forceinline__ v8f wmma_bf16(const V16& a, const V16& b, v8f c) {
  // D = A(16x32 bf16) x B(32x16 bf16) + C(16x16 f32)
  return __builtin_amdgcn_wmma_f32_16x16x32_bf16(
      false, a.v, false, b.v, (short)0, c, false, false);
}

// Low 32 bits of a generic pointer to a __shared__ object are the LDS byte
// address (flat->LDS aperture mapping truncates to addr[31:0]).
__device__ __forceinline__ u32 lds_off(const void* p) {
  return (u32)(uintptr_t)p;
}

// CDNA5 async copy: 16B per lane, global -> LDS, tracked by ASYNCcnt.
__device__ __forceinline__ void async16(u32 lds, const void* g) {
  asm volatile("global_load_async_to_lds_b128 %0, %1, off"
               :: "v"(lds), "v"((u64)(uintptr_t)g) : "memory");
}
__device__ __forceinline__ void wait_async0() {
  asm volatile("s_wait_asynccnt 0x0" ::: "memory");
}
__device__ __forceinline__ void wait_async4() {
  asm volatile("s_wait_asynccnt 0x4" ::: "memory");
}

// ---------------------------------------------------------------- fp32 -> bf16
__global__ void cvt_kernel(const float* __restrict__ in, u16* __restrict__ out, int n) {
  int i = blockIdx.x * 256 + threadIdx.x;
  if (i < n) out[i] = f2bf(in[i]);
}

// ---------------------------------------------------------------- QKV GEMM
// X[8192,1024] bf16, W[3072,1024] bf16 (row = out feature), out = X @ W^T + b
// Double-buffered async-to-LDS staging; epilogue scatters into [B,H,T,HD] Q/K/V.
__global__ __launch_bounds__(256) void qkv_gemm_kernel(
    const u16* __restrict__ X, const u16* __restrict__ W,
    const float* __restrict__ bias,
    u16* __restrict__ qbuf, u16* __restrict__ kbuf, u16* __restrict__ vbuf)
{
  __shared__ __attribute__((aligned(16))) u16 As[2][128 * 40];
  __shared__ __attribute__((aligned(16))) u16 Bs[2][128 * 40];

  const int tid  = threadIdx.x;
  const int lane = tid & 31;
  const int ln   = lane & 15;
  const int hi   = lane >> 4;
  const int wv   = tid >> 5;   // 8 waves
  const int wm   = wv & 3;     // 4 M-quadrants of 32 rows
  const int wn   = wv >> 2;    // 2 N-halves of 64 cols
  const int m0   = blockIdx.y * 128;
  const int n0   = blockIdx.x * 128;

  const int lr = tid >> 1;          // staging row 0..127
  const int lc = (tid & 1) * 16;    // staging col 0 / 16

  const u16* gA = X + (size_t)(m0 + lr) * 1024 + lc;
  const u16* gB = W + (size_t)(n0 + lr) * 1024 + lc;
  u32 la[2], lb[2];
  la[0] = lds_off(&As[0][lr * 40 + lc]); la[1] = lds_off(&As[1][lr * 40 + lc]);
  lb[0] = lds_off(&Bs[0][lr * 40 + lc]); lb[1] = lds_off(&Bs[1][lr * 40 + lc]);

  auto issue = [&](int buf, int k0) {
    async16(la[buf],      gA + k0);
    async16(la[buf] + 16, gA + k0 + 8);
    async16(lb[buf],      gB + k0);
    async16(lb[buf] + 16, gB + k0 + 8);
  };

  const v8f vz = {0.f,0.f,0.f,0.f,0.f,0.f,0.f,0.f};
  v8f acc[2][4];
  for (int i = 0; i < 2; ++i) for (int j = 0; j < 4; ++j) acc[i][j] = vz;

  issue(0, 0);
  for (int it = 0; it < 32; ++it) {
    if (it + 1 < 32) { issue((it + 1) & 1, (it + 1) * 32); wait_async4(); }
    else             { wait_async0(); }
    __syncthreads();

    const u16* as = As[it & 1];
    const u16* bs = Bs[it & 1];
    V16 af[2], bfr[4];
    for (int mi = 0; mi < 2; ++mi) {
      const int m = wm * 32 + mi * 16 + ln;
      af[mi].h[0] = *(const v8bf*)(&as[m * 40 + 8 * hi]);
      af[mi].h[1] = *(const v8bf*)(&as[m * 40 + 16 + 8 * hi]);
    }
    for (int ni = 0; ni < 4; ++ni) {
      const int n = wn * 64 + ni * 16 + ln;
      bfr[ni].h[0] = *(const v8bf*)(&bs[n * 40 + 8 * hi]);
      bfr[ni].h[1] = *(const v8bf*)(&bs[n * 40 + 16 + 8 * hi]);
    }
    for (int mi = 0; mi < 2; ++mi)
      for (int ni = 0; ni < 4; ++ni)
        acc[mi][ni] = wmma_bf16(af[mi], bfr[ni], acc[mi][ni]);
    __syncthreads();
  }

  for (int mi = 0; mi < 2; ++mi)
    for (int ni = 0; ni < 4; ++ni) {
      const int col   = n0 + wn * 64 + ni * 16 + ln;   // 0..3071
      const int which = col >> 10;
      const int c     = col & 1023;
      const int hh    = c >> 6, d = c & 63;
      u16* dst = (which == 0) ? qbuf : (which == 1) ? kbuf : vbuf;
      const float bv = bias[col];
      for (int r = 0; r < 8; ++r) {
        const int row = m0 + wm * 32 + mi * 16 + hi * 8 + r;  // 0..8191
        const int bb = row >> 11, t = row & 2047;
        const size_t off = (((size_t)(bb * 16 + hh)) * 2048 + t) * 64 + d;
        dst[off] = f2bf(acc[mi][ni][r] + bv);
      }
    }
}

// ---------------------------------------------------------------- flash attention
// One workgroup per (qt, h, b): 128 query rows; each wave owns 16 rows.
__global__ __launch_bounds__(256) void attn_kernel(
    const u16* __restrict__ qbuf, const u16* __restrict__ kbuf,
    const u16* __restrict__ vbuf, u16* __restrict__ ao)
{
  __shared__ __attribute__((aligned(16))) u16 Ks[64 * 72];     // [key][d]
  __shared__ __attribute__((aligned(16))) u16 Vt[64 * 72];     // [d][key]
  __shared__ __attribute__((aligned(16))) u16 Ps[8][16 * 72];  // per-wave P tile [m][key]

  const int tid  = threadIdx.x;
  const int lane = tid & 31;
  const int ln   = lane & 15;
  const int hi   = lane >> 4;
  const int wv   = tid >> 5;

  const int qt = blockIdx.x;
  const int hh = blockIdx.y;
  const int bb = blockIdx.z;

  const size_t head = ((size_t)bb * 16 + hh) * 2048 * 64;
  const int q0 = qt * 128 + wv * 16;

  // Q A-fragments: rows q0+ln, head dim split into two K=32 steps
  V16 qf[2];
  {
    const size_t qrow = head + (size_t)(q0 + ln) * 64;
    for (int ks = 0; ks < 2; ++ks) {
      V8 c0, c1;
      c0.u = *(const v4u*)(qbuf + qrow + ks * 32 + 8 * hi);
      c1.u = *(const v4u*)(qbuf + qrow + ks * 32 + 16 + 8 * hi);
      qf[ks].h[0] = c0.b; qf[ks].h[1] = c1.b;
    }
  }

  const v8f vz = {0.f,0.f,0.f,0.f,0.f,0.f,0.f,0.f};
  v8f o[4]; for (int i = 0; i < 4; ++i) o[i] = vz;
  float mrow[8], lrow[8];
  for (int r = 0; r < 8; ++r) { mrow[r] = -1e30f; lrow[r] = 0.f; }

  const int nkb = (qt + 1) * 2;  // causal: only key blocks up to our last row
  for (int kb = 0; kb < nkb; ++kb) {
    const int key0 = kb * 64;
    __syncthreads();
    { // stage K via async copies; V through VGPRs (transposed into LDS)
      const int kr = tid >> 2;
      const int ch = (tid & 3) * 16;
      const size_t src = head + (size_t)(key0 + kr) * 64 + ch;
      async16(lds_off(&Ks[kr * 72 + ch]),     kbuf + src);
      async16(lds_off(&Ks[kr * 72 + ch + 8]), kbuf + src + 8);
      V8 v0v, v1v;
      v0v.u = *(const v4u*)(vbuf + src);
      v1v.u = *(const v4u*)(vbuf + src + 8);
      for (int j = 0; j < 8; ++j) {
        Vt[(ch + j) * 72 + kr]     = v0v.s[j];
        Vt[(ch + 8 + j) * 72 + kr] = v1v.s[j];
      }
    }
    wait_async0();
    __syncthreads();

    // S = Q K^T for four 16-key sub-tiles
    v8f s[4];
    for (int nt = 0; nt < 4; ++nt) {
      v8f sa = vz;
      const int keyl = nt * 16 + ln;
      for (int ks = 0; ks < 2; ++ks) {
        V16 bf;
        bf.h[0] = *(const v8bf*)(&Ks[keyl * 72 + ks * 32 + 8 * hi]);
        bf.h[1] = *(const v8bf*)(&Ks[keyl * 72 + ks * 32 + 16 + 8 * hi]);
        sa = wmma_bf16(qf[ks], bf, sa);
      }
      s[nt] = sa;
    }

    // scale + causal mask + online softmax (C-layout: lane ln = key col, rows 8*hi+r)
    float rmax[8];
    for (int r = 0; r < 8; ++r) rmax[r] = -1e30f;
    for (int nt = 0; nt < 4; ++nt) {
      const int key = key0 + nt * 16 + ln;
      for (int r = 0; r < 8; ++r) {
        const int qrow = q0 + hi * 8 + r;
        float sv = s[nt][r] * 0.125f;              // HD^-0.5
        sv = (key <= qrow) ? sv : -1e30f;
        s[nt][r] = sv;
        rmax[r] = fmaxf(rmax[r], sv);
      }
    }
    for (int r = 0; r < 8; ++r) {
      for (int msk = 1; msk < 16; msk <<= 1)
        rmax[r] = fmaxf(rmax[r], __shfl_xor(rmax[r], msk, 32));
      const float nm = fmaxf(mrow[r], rmax[r]);
      const float sc = __expf(mrow[r] - nm);
      mrow[r] = nm;
      lrow[r] *= sc;
      o[0][r] *= sc; o[1][r] *= sc; o[2][r] *= sc; o[3][r] *= sc;
    }
    float rsum[8];
    for (int r = 0; r < 8; ++r) rsum[r] = 0.f;
    for (int nt = 0; nt < 4; ++nt)
      for (int r = 0; r < 8; ++r) {
        const float p = __expf(s[nt][r] - mrow[r]);
        rsum[r] += p;
        Ps[wv][(hi * 8 + r) * 72 + nt * 16 + ln] = f2bf(p);
      }
    for (int r = 0; r < 8; ++r) {
      for (int msk = 1; msk < 16; msk <<= 1)
        rsum[r] += __shfl_xor(rsum[r], msk, 32);
      lrow[r] += rsum[r];
    }

    // O += P V  (P reloaded from own-wave LDS in A-layout, V from transposed tile)
    for (int dt = 0; dt < 4; ++dt) {
      const int d = dt * 16 + ln;
      for (int ks = 0; ks < 2; ++ks) {
        V16 pa, vb_;
        pa.h[0]  = *(const v8bf*)(&Ps[wv][ln * 72 + ks * 32 + 8 * hi]);
        pa.h[1]  = *(const v8bf*)(&Ps[wv][ln * 72 + ks * 32 + 16 + 8 * hi]);
        vb_.h[0] = *(const v8bf*)(&Vt[d * 72 + ks * 32 + 8 * hi]);
        vb_.h[1] = *(const v8bf*)(&Vt[d * 72 + ks * 32 + 16 + 8 * hi]);
        o[dt] = wmma_bf16(pa, vb_, o[dt]);
      }
    }
  }

  // normalize and store [B,T,H*HD] bf16
  for (int dt = 0; dt < 4; ++dt) {
    const int d = dt * 16 + ln;
    for (int r = 0; r < 8; ++r) {
      const int t = q0 + hi * 8 + r;
      const float val = o[dt][r] / lrow[r];
      ao[((size_t)(bb * 2048 + t)) * 1024 + hh * 64 + d] = f2bf(val);
    }
  }
}

// ---------------------------------------------------------------- output projection
__global__ __launch_bounds__(256) void oproj_gemm_kernel(
    const u16* __restrict__ X, const u16* __restrict__ W,
    const float* __restrict__ bias, float* __restrict__ out)
{
  __shared__ __attribute__((aligned(16))) u16 As[2][128 * 40];
  __shared__ __attribute__((aligned(16))) u16 Bs[2][128 * 40];

  const int tid  = threadIdx.x;
  const int lane = tid & 31;
  const int ln   = lane & 15;
  const int hi   = lane >> 4;
  const int wv   = tid >> 5;
  const int wm   = wv & 3;
  const int wn   = wv >> 2;
  const int m0   = blockIdx.y * 128;
  const int n0   = blockIdx.x * 128;

  const int lr = tid >> 1;
  const int lc = (tid & 1) * 16;

  const u16* gA = X + (size_t)(m0 + lr) * 1024 + lc;
  const u16* gB = W + (size_t)(n0 + lr) * 1024 + lc;
  u32 la[2], lb[2];
  la[0] = lds_off(&As[0][lr * 40 + lc]); la[1] = lds_off(&As[1][lr * 40 + lc]);
  lb[0] = lds_off(&Bs[0][lr * 40 + lc]); lb[1] = lds_off(&Bs[1][lr * 40 + lc]);

  auto issue = [&](int buf, int k0) {
    async16(la[buf],      gA + k0);
    async16(la[buf] + 16, gA + k0 + 8);
    async16(lb[buf],      gB + k0);
    async16(lb[buf] + 16, gB + k0 + 8);
  };

  const v8f vz = {0.f,0.f,0.f,0.f,0.f,0.f,0.f,0.f};
  v8f acc[2][4];
  for (int i = 0; i < 2; ++i) for (int j = 0; j < 4; ++j) acc[i][j] = vz;

  issue(0, 0);
  for (int it = 0; it < 32; ++it) {
    if (it + 1 < 32) { issue((it + 1) & 1, (it + 1) * 32); wait_async4(); }
    else             { wait_async0(); }
    __syncthreads();

    const u16* as = As[it & 1];
    const u16* bs = Bs[it & 1];
    V16 af[2], bfr[4];
    for (int mi = 0; mi < 2; ++mi) {
      const int m = wm * 32 + mi * 16 + ln;
      af[mi].h[0] = *(const v8bf*)(&as[m * 40 + 8 * hi]);
      af[mi].h[1] = *(const v8bf*)(&as[m * 40 + 16 + 8 * hi]);
    }
    for (int ni = 0; ni < 4; ++ni) {
      const int n = wn * 64 + ni * 16 + ln;
      bfr[ni].h[0] = *(const v8bf*)(&bs[n * 40 + 8 * hi]);
      bfr[ni].h[1] = *(const v8bf*)(&bs[n * 40 + 16 + 8 * hi]);
    }
    for (int mi = 0; mi < 2; ++mi)
      for (int ni = 0; ni < 4; ++ni)
        acc[mi][ni] = wmma_bf16(af[mi], bfr[ni], acc[mi][ni]);
    __syncthreads();
  }

  for (int mi = 0; mi < 2; ++mi)
    for (int ni = 0; ni < 4; ++ni) {
      const int col = n0 + wn * 64 + ni * 16 + ln;
      const float bv = bias[col];
      for (int r = 0; r < 8; ++r) {
        const int row = m0 + wm * 32 + mi * 16 + hi * 8 + r;
        out[(size_t)row * 1024 + col] = acc[mi][ni][r] + bv;
      }
    }
}

// ---------------------------------------------------------------- launcher
extern "C" void kernel_launch(void* const* d_in, const int* in_sizes, int n_in,
                              void* d_out, int out_size, void* d_ws, size_t ws_size,
                              hipStream_t stream) {
  const float* hs    = (const float*)d_in[0];   // [4,2048,1024]
  const float* qkv_w = (const float*)d_in[1];   // [3072,1024]
  const float* qkv_b = (const float*)d_in[2];   // [3072]
  const float* o_w   = (const float*)d_in[3];   // [1024,1024]
  const float* o_b   = (const float*)d_in[4];   // [1024]
  float* out = (float*)d_out;

  u16* ws   = (u16*)d_ws;
  u16* hsb  = ws;                                   // 8192*1024 bf16
  u16* wqkv = hsb  + (size_t)8192 * 1024;           // 3072*1024
  u16* wo   = wqkv + (size_t)3072 * 1024;           // 1024*1024
  u16* qb   = wo   + (size_t)1024 * 1024;           // 4*16*2048*64
  u16* kb   = qb   + (size_t)4 * 16 * 2048 * 64;
  u16* vb   = kb   + (size_t)4 * 16 * 2048 * 64;
  u16* ao   = vb   + (size_t)4 * 16 * 2048 * 64;    // 8192*1024

  cvt_kernel<<<(8192 * 1024) / 256, 256, 0, stream>>>(hs, hsb, 8192 * 1024);
  cvt_kernel<<<(3072 * 1024) / 256, 256, 0, stream>>>(qkv_w, wqkv, 3072 * 1024);
  cvt_kernel<<<(1024 * 1024) / 256, 256, 0, stream>>>(o_w, wo, 1024 * 1024);

  qkv_gemm_kernel<<<dim3(24, 64), 256, 0, stream>>>(hsb, wqkv, qkv_b, qb, kb, vb);
  attn_kernel<<<dim3(16, 16, 4), 256, 0, stream>>>(qb, kb, vb, ao);
  oproj_gemm_kernel<<<dim3(8, 64), 256, 0, stream>>>(ao, wo, o_b, out);
}
